// RADS_72670846648856
// MI455X (gfx1250) — matrix operand order repaired and checked by reading, not compiled, not run on hardware
//
#include <hip/hip_runtime.h>
#include <hip/hip_bf16.h>
#include <math.h>

// ---------------------------------------------------------------------------
// Problem: B=256, N=262144, D=256, E=64, H=256
// outputs: logits [B] then attn [B*N], concatenated flat in d_out (f32).
// ---------------------------------------------------------------------------

#define BQ   256
#define NTOT 262144
#define DD   256
#define EE   64
#define HH   256
#define NEG_INF (-1e30f)

typedef __attribute__((ext_vector_type(16))) __bf16 v16bf;
typedef __attribute__((ext_vector_type(2)))  __bf16 v2bf;
typedef __attribute__((ext_vector_type(8)))  float  v8f;

// ---- f32 -> bf16, packed pair. Prefer the native packed cvt instruction. ----
__device__ __forceinline__ v2bf f2bf2(float x, float y) {
#if __has_builtin(__builtin_amdgcn_cvt_pk_bf16_f32)
    return __builtin_amdgcn_cvt_pk_bf16_f32(x, y);
#else
    v2bf r;
    r[0] = (__bf16)x;   // let the backend select its native v_cvt
    r[1] = (__bf16)y;
    return r;
#endif
}

// ---------------------------------------------------------------------------
// K0: zero the context accumulator in workspace (ws is poisoned 0xAA).
// ---------------------------------------------------------------------------
__global__ void k_zero(float* __restrict__ p, int n) {
    int i = blockIdx.x * blockDim.x + threadIdx.x;
    if (i < n) p[i] = 0.0f;
}

// ---------------------------------------------------------------------------
// K1: raw masked scores = q @ (lib/tau)^T  -> written into attn buffer.
// grid.x = N/128; block = 256 threads = 8 waves; each wave owns a 16-col
// tile of N, preloads its bf16 B-fragments once (with 1/tau folded in),
// then loops all 16 M-tiles. The library is read from HBM exactly once;
// q comes from L2. The inner loop is pure load+cvt+wmma.
// ---------------------------------------------------------------------------
__global__ void __launch_bounds__(256) k_scores(
    const float* __restrict__ q,
    const float* __restrict__ lib,
    const unsigned char* __restrict__ mask,   // numpy bool = 1 byte
    const float* __restrict__ log_tau,
    float* __restrict__ scores)               // [B, NTOT] raw masked scores
{
    const int tid  = threadIdx.x;
    const int wave = tid >> 5;
    const int lane = tid & 31;
    const int half = lane >> 4;
    const int l16  = lane & 15;
    const int nbase = blockIdx.x * 128 + wave * 16;

    float tau = __expf(log_tau[0]);
    tau = fminf(fmaxf(tau, 0.001f), 10.0f);
    const float invTau = 1.0f / tau;

    // Preload B fragments: B[k][n] = lib[nbase+n][k] * invTau  (scale folded
    // into B so the per-M-tile A path is load+cvt only).
    // Layout (bf16 32x16 B): dword v holds K pair base=(v<4?0:16)+(lane<16?0:8), K=base+(v%4)*2 ; N = lane%16.
    v16bf Bf[8];
    const float* bbase = lib + (size_t)(nbase + l16) * DD;
#pragma unroll
    for (int ks = 0; ks < 8; ++ks) {
#pragma unroll
        for (int v = 0; v < 8; ++v) {
            int kB = ks * 32 + ((v < 4) ? 0 : 16) + (half ? 8 : 0) + (v & 3) * 2;
            float2 fb = *(const float2*)(bbase + kB);
            v2bf pk = f2bf2(fb.x * invTau, fb.y * invTau);
            Bf[ks][2 * v]     = pk[0];
            Bf[ks][2 * v + 1] = pk[1];
        }
    }

    for (int mt = 0; mt < 16; ++mt) {
        const int mtile = mt * 16;
        const float* abase = q + (size_t)(mtile + l16) * DD;
        v8f acc = {};
#pragma unroll
        for (int ks = 0; ks < 8; ++ks) {
            // A layout (bf16 16x32): M = lane%16; dword v holds K pair:
            // base = (v<4 ? 0 : 16) + (lane<16 ? 0 : 8), K = base + (v%4)*2.
            v16bf a;
#pragma unroll
            for (int v = 0; v < 8; ++v) {
                int kA = ks * 32 + ((v < 4) ? 0 : 16) + (half ? 8 : 0) + (v & 3) * 2;
                float2 fa = *(const float2*)(abase + kA);
                v2bf pk = f2bf2(fa.x, fa.y);
                a[2 * v]     = pk[0];
                a[2 * v + 1] = pk[1];
            }
            acc = __builtin_amdgcn_wmma_f32_16x16x32_bf16(
                false, a, false, Bf[ks], (short)0, acc, false, false);
        }
        // C layout: element (vgpr r, lane): M = r + (lane<16?0:8), N = lane%16.
#pragma unroll
        for (int r = 0; r < 8; ++r) {
            int m = mtile + r + (half ? 8 : 0);
            size_t idx = (size_t)m * NTOT + nbase + l16;
            float s = acc[r];
            if (mask[idx]) s = NEG_INF;
            scores[idx] = s;
        }
    }
}

// ---------------------------------------------------------------------------
// K2: per-row online softmax stats (max, 1/sumexp). One block per row.
// ---------------------------------------------------------------------------
__global__ void __launch_bounds__(256) k_rowstats(
    const float* __restrict__ scores,
    float* __restrict__ rowMax,
    float* __restrict__ rowInvSum)
{
    __shared__ float sm[256], ss[256];
    const int b = blockIdx.x, t = threadIdx.x;
    const float* row = scores + (size_t)b * NTOT;
    float m = -3.0e38f, s = 0.0f;
    for (int i = t; i < NTOT; i += 256) {
        float x = row[i];
        if (x > m) { s *= __expf(m - x); m = x; }
        s += __expf(x - m);
    }
    sm[t] = m; ss[t] = s;
    __syncthreads();
    for (int off = 128; off > 0; off >>= 1) {
        if (t < off) {
            float m2 = sm[t + off], s2 = ss[t + off];
            float M  = fmaxf(sm[t], m2);
            ss[t] = ss[t] * __expf(sm[t] - M) + s2 * __expf(m2 - M);
            sm[t] = M;
        }
        __syncthreads();
    }
    if (t == 0) { rowMax[b] = sm[0]; rowInvSum[b] = 1.0f / ss[0]; }
}

// ---------------------------------------------------------------------------
// K3: normalize scores -> attn (final output) and accumulate
// context += attn @ lib with bf16 WMMA (split-K over N, f32 atomics).
// grid = (16 M-tiles, 128 N-chunks of 2048); M fastest so adjacent blocks
// reuse the same 2 MB library chunk from L2.
// ---------------------------------------------------------------------------
__global__ void __launch_bounds__(256) k_ctx(
    const float* __restrict__ lib,
    const float* __restrict__ rowMax,
    const float* __restrict__ rowInvSum,
    float* __restrict__ attn,                 // in: raw scores, out: probs
    float* __restrict__ ctx)                  // [B, D] f32 accumulator (zeroed)
{
    __shared__ float pLds[16 * 128];          // 8 KB probability slab
    const int tid  = threadIdx.x;
    const int wave = tid >> 5;
    const int lane = tid & 31;
    const int half = lane >> 4;
    const int l16  = lane & 15;
    const int mtile  = blockIdx.x * 16;
    const int nchunk = blockIdx.y * 2048;
    const int d0 = wave * 16;                 // each wave owns two D-subtiles
    const int d1 = (wave + 8) * 16;

    v8f c0 = {}, c1 = {};
    for (int slab = 0; slab < 16; ++slab) {
        const int nslab = nchunk + slab * 128;
        __syncthreads();
        // normalize 16x128 slab: write final attn + stage probs in LDS
#pragma unroll
        for (int k = 0; k < 8; ++k) {
            int j = k * 256 + tid;
            int row = j >> 7, col = j & 127;
            int gm = mtile + row;
            size_t gi = (size_t)gm * NTOT + nslab + col;
            float p = __expf(attn[gi] - rowMax[gm]) * rowInvSum[gm];
            pLds[row * 128 + col] = p;
            attn[gi] = p;
        }
        __syncthreads();
#pragma unroll
        for (int ks = 0; ks < 4; ++ks) {
            const int koff = ks * 32;
            v16bf a;
#pragma unroll
            for (int v = 0; v < 8; ++v) {
                int kA = koff + ((v < 4) ? 0 : 16) + (half ? 8 : 0) + (v & 3) * 2;
                float2 fa = *(const float2*)&pLds[l16 * 128 + kA];
                v2bf pk = f2bf2(fa.x, fa.y);
                a[2 * v]     = pk[0];
                a[2 * v + 1] = pk[1];
            }
            v16bf b0, b1;
#pragma unroll
            for (int v = 0; v < 8; ++v) {
                int kB = koff + ((v < 4) ? 0 : 16) + (half ? 8 : 0) + (v & 3) * 2;
                const float* bp0 = lib + (size_t)(nslab + kB) * DD + d0 + l16;
                v2bf p0 = f2bf2(bp0[0], bp0[DD]);
                b0[2 * v]     = p0[0];
                b0[2 * v + 1] = p0[1];
                const float* bp1 = lib + (size_t)(nslab + kB) * DD + d1 + l16;
                v2bf p1 = f2bf2(bp1[0], bp1[DD]);
                b1[2 * v]     = p1[0];
                b1[2 * v + 1] = p1[1];
            }
            c0 = __builtin_amdgcn_wmma_f32_16x16x32_bf16(
                false, a, false, b0, (short)0, c0, false, false);
            c1 = __builtin_amdgcn_wmma_f32_16x16x32_bf16(
                false, a, false, b1, (short)0, c1, false, false);
        }
    }
#pragma unroll
    for (int r = 0; r < 8; ++r) {
        int gm = mtile + r + (half ? 8 : 0);
        atomicAdd(&ctx[gm * DD + d0 + l16], c0[r]);
        atomicAdd(&ctx[gm * DD + d1 + l16], c1[r]);
    }
}

// ---------------------------------------------------------------------------
// K4: head MLP. h = gelu([q, context, action] @ w1 + b1); logit = h @ w2 + b2.
// One block per batch row, one thread per hidden unit.
// ---------------------------------------------------------------------------
__global__ void __launch_bounds__(256) k_head(
    const float* __restrict__ q,
    const float* __restrict__ act,
    const float* __restrict__ ctx,
    const float* __restrict__ w1,             // [2D+E, H]
    const float* __restrict__ b1,             // [H]
    const float* __restrict__ w2,             // [H, 1]
    const float* __restrict__ b2,             // [1]
    float* __restrict__ logits)               // [B]
{
    __shared__ float red[256];
    const int b = blockIdx.x, j = threadIdx.x;
    float acc = b1[j];
    for (int i = 0; i < DD; ++i) acc += q[b * DD + i]   * w1[i * HH + j];
    for (int i = 0; i < DD; ++i) acc += ctx[b * DD + i] * w1[(DD + i) * HH + j];
    for (int i = 0; i < EE; ++i) acc += act[b * EE + i] * w1[(2 * DD + i) * HH + j];
    // jax.nn.gelu default = tanh approximation
    float x = acc;
    float h = 0.5f * x * (1.0f + tanhf(0.7978845608028654f * (x + 0.044715f * x * x * x)));
    red[j] = h * w2[j];
    __syncthreads();
    for (int off = 128; off > 0; off >>= 1) {
        if (j < off) red[j] += red[j + off];
        __syncthreads();
    }
    if (j == 0) logits[b] = red[0] + b2[0];
}

// ---------------------------------------------------------------------------
extern "C" void kernel_launch(void* const* d_in, const int* in_sizes, int n_in,
                              void* d_out, int out_size, void* d_ws, size_t ws_size,
                              hipStream_t stream)
{
    (void)in_sizes; (void)n_in; (void)out_size; (void)ws_size;

    const float*         q       = (const float*)d_in[0];
    const float*         act     = (const float*)d_in[1];
    const float*         lib     = (const float*)d_in[2];
    const unsigned char* mask    = (const unsigned char*)d_in[3]; // jnp.bool_ -> 1 byte
    const float*         w1      = (const float*)d_in[4];
    const float*         b1      = (const float*)d_in[5];
    const float*         w2      = (const float*)d_in[6];
    const float*         b2      = (const float*)d_in[7];
    const float*         log_tau = (const float*)d_in[8];

    float* out    = (float*)d_out;
    float* logits = out;            // [256]
    float* attn   = out + BQ;       // [256 * 262144]

    float* ws        = (float*)d_ws;
    float* rowMax    = ws;          // [256]
    float* rowInvSum = ws + 256;    // [256]
    float* ctx       = ws + 512;    // [256*256] -> total ws use = 264 KB

    k_zero<<<(BQ * DD + 255) / 256, 256, 0, stream>>>(ctx, BQ * DD);
    k_scores<<<NTOT / 128, 256, 0, stream>>>(q, lib, mask, log_tau, attn);
    k_rowstats<<<BQ, 256, 0, stream>>>(attn, rowMax, rowInvSum);
    k_ctx<<<dim3(BQ / 16, NTOT / 2048), 256, 0, stream>>>(lib, rowMax, rowInvSum, attn, ctx);
    k_head<<<BQ, 256, 0, stream>>>(q, act, ctx, w1, b1, w2, b2, logits);
}
